// Detect_46437186404385
// MI455X (gfx1250) — matrix operand order repaired and checked by reading, not compiled, not run on hardware
//
#include <hip/hip_runtime.h>
#include <hip/hip_bf16.h>
#include <math.h>

typedef _Float16 v16h __attribute__((ext_vector_type(16)));
typedef _Float16 v8h  __attribute__((ext_vector_type(8)));
typedef float    v8f  __attribute__((ext_vector_type(8)));

#define BSZ 8  // batch size

// ------------------------------------------------------------------
// fp32 NCHW -> fp16 N,C,(H+2),(W+2) with explicit zero border.
// Writing the zeros every call keeps the buffer deterministic.
// ------------------------------------------------------------------
__global__ void k_cvt_pad(const float* __restrict__ src, _Float16* __restrict__ dst,
                          int H, int W, int total) {
    int i = blockIdx.x * blockDim.x + threadIdx.x;
    if (i >= total) return;
    const int Wp = W + 2, PHW = (H + 2) * Wp;
    const int nc = i / PHW;
    const int r  = i % PHW;
    const int yp = r / Wp, xp = r % Wp;
    _Float16 v = (_Float16)0.f;
    if (yp >= 1 && yp <= H && xp >= 1 && xp <= W)
        v = (_Float16)src[(size_t)nc * H * W + (yp - 1) * W + (xp - 1)];
    dst[i] = v;
}

// ------------------------------------------------------------------
// Repack 3x3 conv weights (OIHW fp32) into the WMMA A-matrix per-lane
// streaming order (16x32 A tile, contiguous k order as in the B reads):
//   lane L (group g = L>>4, M = L&15), element e in v16h:
//     K = e + (g<<4)
// TAP-FASTEST stream: (((tile*NC + cc)*9 + tap)*32 + lane)*16 + e
// -> the 9 taps of one K chunk are 9 consecutive 1KB lines (imm offsets).
// ------------------------------------------------------------------
__global__ void k_repack_w3(const float* __restrict__ w, _Float16* __restrict__ out, int C) {
    int idx = blockIdx.x * blockDim.x + threadIdx.x;
    int total = C * C * 9;
    if (idx >= total) return;
    int e    = idx & 15;
    int lane = (idx >> 4) & 31;
    int r    = idx >> 9;           // (tile*NC + cc)*9 + tap
    int nc   = C >> 5;
    int tap  = r % 9;
    int r2   = r / 9;
    int cc   = r2 % nc;
    int tile = r2 / nc;
    int g = lane >> 4, M = lane & 15;
    int K = e + (g << 4);
    int co = tile * 16 + M;
    int ci = cc * 32 + K;
    out[idx] = (_Float16)w[(co * C + ci) * 9 + tap];
}

// 1x1 conv weights (Mv x C fp32), padded to 32 rows (2 M-tiles of 16).
__global__ void k_repack_w1(const float* __restrict__ w, _Float16* __restrict__ out,
                            int C, int Mv) {
    int idx = blockIdx.x * blockDim.x + threadIdx.x;
    int total = 32 * C;
    if (idx >= total) return;
    int e    = idx & 15;
    int lane = (idx >> 4) & 31;
    int rest = idx >> 9;           // tile*(C/32) + cc
    int nc   = C >> 5;
    int cc   = rest % nc;
    int tile = rest / nc;
    int g = lane >> 4, M = lane & 15;
    int K = e + (g << 4);
    int co = tile * 16 + M;
    int ci = cc * 32 + K;
    out[idx] = (co < Mv) ? (_Float16)w[co * C + ci] : (_Float16)0.f;
}

// ------------------------------------------------------------------
// Implicit-GEMM 3x3 conv on the zero-padded image, all-WMMA inner loop.
//
// Block = 256 threads = 8 wave32s arranged 4(M) x 2(N); each wave owns a
// 2(Mtile) x 2(Ncol) register tile (4 v8f accumulators) -> block tile
// 128co x 64pos, one image per block (blockIdx.z).
//
// Per 32-channel chunk the flat halo of the padded image is staged ONCE
// into LDS (pure copy, no branches, no selects -- borders are real zeros)
// and all 9 taps read it at compile-time LDS offsets. A comes from the
// tap-fastest packed stream at immediate offsets. LDS double-buffered:
// one barrier per chunk, 36 WMMAs per wave per chunk.
// ------------------------------------------------------------------
template <int C, int H, int W>
__global__ __launch_bounds__(256) void k_conv3x3_wmma(
    const _Float16* __restrict__ in,      // padded (H+2)x(W+2)
    const _Float16* __restrict__ Aw,
    const float* __restrict__ bias,
    _Float16* __restrict__ out) {         // unpadded HxW
    constexpr int HW  = H * W;
    constexpr int Wp  = W + 2;
    constexpr int PHW = (H + 2) * Wp;
    constexpr int NC  = C >> 5;
    constexpr int TN  = 64;
    constexpr int WRAPS = 63 / W + 1;
    constexpr int F   = TN + 2 * WRAPS + 2 * Wp + 4;  // halo length (padded flat)
    constexpr int FS  = 40;                           // LDS stride (halves)
    constexpr int FQ  = (F + 31) / 32;

    __shared__ __align__(16) _Float16 Bs[2 * F * FS];

    const int tid  = threadIdx.x;
    const int wave = tid >> 5;
    const int lane = tid & 31;
    const int wvM  = wave >> 1;          // 0..3 -> co tiles 2*wvM, 2*wvM+1
    const int wvN  = wave & 1;           // 0..1 -> cols wvN*32 + {0,16}
    const int posBlk = blockIdx.x * TN;
    const int coBlk  = blockIdx.y * 128;
    const int n      = blockIdx.z;

    // halo window in padded flat coords
    const int y0 = posBlk / W, x0 = posBlk % W;
    const int baseBlk   = (y0 + 1) * Wp + (x0 + 1);
    const int haloStart = baseBlk - Wp - 1;

    // this wave's two output columns (per-lane)
    const int local0 = wvN * 32 + (lane & 15);
    const int local1 = local0 + 16;
    const int pos0 = posBlk + local0, pos1 = posBlk + local1;
    // padded flat base of each column, relative to halo start (>= 0)
    const int fOff0 = (pos0 / W + 1) * Wp + pos0 % W + 1 - haloStart;
    const int fOff1 = (pos1 / W + 1) * Wp + pos1 % W + 1 - haloStart;

    const int tileA0 = (coBlk >> 4) + wvM * 2;
    const int tileA1 = tileA0 + 1;
    const size_t chanInBase  = (size_t)n * C * PHW;
    const size_t chanOutBase = (size_t)n * C * HW;

    v8f acc00 = {}, acc01 = {}, acc10 = {}, acc11 = {};

    auto stage = [&](int cc, int buf) {
        const _Float16* src = in + chanInBase + (size_t)cc * 32 * PHW;
        _Float16* dst = &Bs[buf * (F * FS)];
#pragma unroll
        for (int kq = 0; kq < 4; ++kq) {
            const int k = kq * 8 + wave;
            const _Float16* sk = src + (size_t)k * PHW;
#pragma unroll
            for (int fq = 0; fq < FQ; ++fq) {
                const int f = fq * 32 + lane;
                if (f < F) {
                    int flat = haloStart + f;
                    flat = flat < PHW ? flat : PHW - 1;   // tail-of-image clamp only
                    dst[f * FS + k] = sk[flat];
                }
            }
        }
    };

    auto compute = [&](int cc, int buf) {
        const _Float16* Bb = &Bs[buf * (F * FS)];
        const int gOff = (lane >> 4) << 4;
        const _Float16* A0 = Aw + ((size_t)(tileA0 * NC + cc) * 9) * 512 + lane * 16;
        const _Float16* A1 = Aw + ((size_t)(tileA1 * NC + cc) * 9) * 512 + lane * 16;
        __builtin_prefetch(A0 + 9 * 512, 0, 3);   // next chunk's A stream
        __builtin_prefetch(A1 + 9 * 512, 0, 3);
#pragma unroll
        for (int tap = 0; tap < 9; ++tap) {
            const int dy = tap / 3, dx = tap % 3;         // 0..2 (pre-shifted)
            const int tapF = dy * Wp + dx;                // compile-time constant

            union { v16h v; v8h h[2]; } b0, b1;
            const _Float16* bp0 = Bb + (fOff0 + tapF) * FS + gOff;
            const _Float16* bp1 = Bb + (fOff1 + tapF) * FS + gOff;
            b0.h[0] = *reinterpret_cast<const v8h*>(bp0);
            b0.h[1] = *reinterpret_cast<const v8h*>(bp0 + 8);
            b1.h[0] = *reinterpret_cast<const v8h*>(bp1);
            b1.h[1] = *reinterpret_cast<const v8h*>(bp1 + 8);

            v16h a0 = *reinterpret_cast<const v16h*>(A0 + tap * 512);
            v16h a1 = *reinterpret_cast<const v16h*>(A1 + tap * 512);

            acc00 = __builtin_amdgcn_wmma_f32_16x16x32_f16(false, a0, false, b0.v, (short)0, acc00, false, false);
            acc01 = __builtin_amdgcn_wmma_f32_16x16x32_f16(false, a0, false, b1.v, (short)0, acc01, false, false);
            acc10 = __builtin_amdgcn_wmma_f32_16x16x32_f16(false, a1, false, b0.v, (short)0, acc10, false, false);
            acc11 = __builtin_amdgcn_wmma_f32_16x16x32_f16(false, a1, false, b1.v, (short)0, acc11, false, false);
        }
    };

    stage(0, 0);
    for (int cc = 0; cc < NC; ++cc) {
        __syncthreads();
        if (cc + 1 < NC) stage(cc + 1, (cc + 1) & 1);
        compute(cc, cc & 1);
    }

    // ---- epilogue: C/D layout -> lanes 0-15: M=j, lanes 16-31: M=j+8 ----
    const int gRow = (lane >> 4) << 3;
    const int coT0 = coBlk + wvM * 32 + gRow;
    const int coT1 = coT0 + 16;
    if (pos0 < HW) {
#pragma unroll
        for (int j = 0; j < 8; ++j) {
            out[chanOutBase + (size_t)(coT0 + j) * HW + pos0] = (_Float16)(acc00[j] + bias[coT0 + j]);
            out[chanOutBase + (size_t)(coT1 + j) * HW + pos0] = (_Float16)(acc10[j] + bias[coT1 + j]);
        }
    }
    if (pos1 < HW) {
#pragma unroll
        for (int j = 0; j < 8; ++j) {
            out[chanOutBase + (size_t)(coT0 + j) * HW + pos1] = (_Float16)(acc01[j] + bias[coT0 + j]);
            out[chanOutBase + (size_t)(coT1 + j) * HW + pos1] = (_Float16)(acc11[j] + bias[coT1 + j]);
        }
    }
}

// ------------------------------------------------------------------
// 1x1 conv (head): M=32 (padded, 2 wave-rows) x N=64 (4 wave-cols).
// ------------------------------------------------------------------
template <int C>
__global__ __launch_bounds__(256) void k_conv1x1_wmma(
    const _Float16* __restrict__ in, const _Float16* __restrict__ Aw,
    const float* __restrict__ bias, float* __restrict__ out, int HW, int Mv) {
    const int NPOS = BSZ * HW;
    constexpr int NC = C >> 5;
    __shared__ __align__(16) _Float16 Bs[64 * 40];

    const int tid  = threadIdx.x;
    const int wave = tid >> 5;
    const int lane = tid & 31;
    const int wrow = wave >> 2;   // 0..1 (M)
    const int wcol = wave & 3;    // 0..3 (N)
    const int posBlk = blockIdx.x * 64;

    const int sCol  = tid & 63;
    const int kBase = tid >> 6;   // 0..3
    const int sPos  = posBlk + sCol;
    const bool sOk  = sPos < NPOS;
    const int sN = sOk ? sPos / HW : 0;
    const int sR = sOk ? sPos % HW : 0;

    v8f acc = {};
    for (int cc = 0; cc < NC; ++cc) {
#pragma unroll
        for (int p = 0; p < 8; ++p) {
            int k = p * 4 + kBase;
            _Float16 v = (_Float16)0.f;
            if (sOk) v = in[(sN * C + cc * 32 + k) * (size_t)HW + sR];
            Bs[sCol * 40 + k] = v;
        }
        __syncthreads();
        const v16h* Ap =
            reinterpret_cast<const v16h*>(Aw + ((size_t)(wrow * NC + cc) << 9));
        v16h a = Ap[lane];
        union { v16h v; v8h h[2]; } b;
        const _Float16* bp = &Bs[(wcol * 16 + (lane & 15)) * 40 + ((lane >> 4) << 4)];
        b.h[0] = *reinterpret_cast<const v8h*>(bp);
        b.h[1] = *reinterpret_cast<const v8h*>(bp + 8);
        acc = __builtin_amdgcn_wmma_f32_16x16x32_f16(
            false, a, false, b.v, (short)0, acc, false, false);
        __syncthreads();
    }
    const int pos = posBlk + wcol * 16 + (lane & 15);
    if (pos < NPOS) {
        const int n = pos / HW, r = pos % HW;
        const int coB = wrow * 16 + ((lane >> 4) << 3);
#pragma unroll
        for (int j = 0; j < 8; ++j) {
            int co = coB + j;
            if (co < Mv) out[(n * Mv + co) * (size_t)HW + r] = acc[j] + bias[co];
        }
    }
}

// ------------------------------------------------------------------
// Decode epilogue (elementwise, bandwidth-bound).
// ------------------------------------------------------------------
__global__ void k_decode(const float* __restrict__ h2, const float* __restrict__ h3,
                         float* __restrict__ out, int HW, int nx, float stride,
                         float a0w, float a0h, float a1w, float a1h,
                         float a2w, float a2h, int scaleOff) {
    const int r = blockIdx.x * blockDim.x + threadIdx.x;
    const int n = BSZ * 3 * HW;
    if (r >= n) return;
    const int b  = r / (3 * HW);
    int rem      = r % (3 * HW);
    const int a  = rem / HW;
    const int pr = rem % HW;
    const int y = pr / nx, x = pr % nx;

    float s[8];
#pragma unroll
    for (int j = 0; j < 8; ++j) {
        float v = h2[(b * 24 + a * 8 + j) * HW + pr];
        s[j] = 1.f / (1.f + expf(-v));
    }
    float u[9];
#pragma unroll
    for (int j = 0; j < 9; ++j) u[j] = h3[(b * 27 + a * 9 + j) * HW + pr];

    const float aw = (a == 0) ? a0w : (a == 1) ? a1w : a2w;
    const float ah = (a == 0) ? a0h : (a == 1) ? a1h : a2h;

    float* o = out + (size_t)(b * 25200 + scaleOff + a * HW + pr) * 17;
    o[0] = (s[0] * 2.f - 0.5f + (float)x) * stride;
    o[1] = (s[1] * 2.f - 0.5f + (float)y) * stride;
    float tw = s[2] * 2.f, th = s[3] * 2.f;
    o[2] = tw * tw * aw;
    o[3] = th * th * ah;
    o[4] = s[4]; o[5] = s[5]; o[6] = s[6]; o[7] = s[7];
    o[8] = u[0]; o[9] = u[1];
    float n1 = fmaxf(sqrtf(u[2] * u[2] + u[3] * u[3]), 1e-12f);
    float n2 = fmaxf(sqrtf(u[4] * u[4] + u[5] * u[5]), 1e-12f);
    o[10] = u[2] / n1; o[11] = u[3] / n1;
    o[12] = u[4] / n2; o[13] = u[5] / n2;
    o[14] = 1.f / (1.f + expf(-u[6])) * 2.f - 1.f;
    o[15] = 1.f / (1.f + expf(-u[7])) * 2.f - 1.f;
    o[16] = 1.f / (1.f + expf(-u[8])) * 2.f - 1.f;
}

// ------------------------------------------------------------------
// Host orchestration
// ------------------------------------------------------------------
extern "C" void kernel_launch(void* const* d_in, const int* in_sizes, int n_in,
                              void* d_out, int out_size, void* d_ws, size_t ws_size,
                              hipStream_t stream) {
    (void)in_sizes; (void)n_in; (void)out_size; (void)ws_size;
    static const int   Cs[3]       = {256, 512, 1024};
    static const int   Hs[3]       = {80, 40, 20};
    static const float strides[3]  = {8.f, 16.f, 32.f};
    static const float anch[3][6]  = {{10, 13, 16, 30, 33, 23},
                                      {30, 61, 62, 45, 59, 119},
                                      {116, 90, 156, 198, 373, 326}};
    static const int   scaleOff[3] = {0, 19200, 24000};

    char* ws = (char*)d_ws;
    size_t cur = 0;
    auto alloc = [&](size_t bytes) -> void* {
        void* p = ws + cur;
        cur = (cur + bytes + 255) & ~(size_t)255;
        return p;
    };

    for (int i = 0; i < 3; ++i) {
        const int C = Cs[i], H = Hs[i], W = Hs[i], HW = H * W;
        const int PHW = (H + 2) * (W + 2);
        const float* f   = (const float*)d_in[i * 9 + 0];
        const float* w2a = (const float*)d_in[i * 9 + 1];
        const float* b2a = (const float*)d_in[i * 9 + 2];
        const float* w2b = (const float*)d_in[i * 9 + 3];
        const float* b2b = (const float*)d_in[i * 9 + 4];
        const float* w3a = (const float*)d_in[i * 9 + 5];
        const float* b3a = (const float*)d_in[i * 9 + 6];
        const float* w3b = (const float*)d_in[i * 9 + 7];
        const float* b3b = (const float*)d_in[i * 9 + 8];

        _Float16* in16 = (_Float16*)alloc((size_t)BSZ * C * PHW * 2);  // padded
        _Float16* Aw2a = (_Float16*)alloc((size_t)C * C * 9 * 2);
        _Float16* Aw3a = (_Float16*)alloc((size_t)C * C * 9 * 2);
        _Float16* Aw2b = (_Float16*)alloc((size_t)32 * C * 2);
        _Float16* Aw3b = (_Float16*)alloc((size_t)32 * C * 2);
        _Float16* int2 = (_Float16*)alloc((size_t)BSZ * C * HW * 2);
        _Float16* int3 = (_Float16*)alloc((size_t)BSZ * C * HW * 2);
        float* h2raw   = (float*)alloc((size_t)BSZ * 24 * HW * 4);
        float* h3raw   = (float*)alloc((size_t)BSZ * 27 * HW * 4);

        const int nPad = BSZ * C * PHW;
        k_cvt_pad<<<(nPad + 255) / 256, 256, 0, stream>>>(f, in16, H, W, nPad);
        const int nW3 = C * C * 9;
        k_repack_w3<<<(nW3 + 255) / 256, 256, 0, stream>>>(w2a, Aw2a, C);
        k_repack_w3<<<(nW3 + 255) / 256, 256, 0, stream>>>(w3a, Aw3a, C);
        const int nW1 = 32 * C;
        k_repack_w1<<<(nW1 + 255) / 256, 256, 0, stream>>>(w2b, Aw2b, C, 24);
        k_repack_w1<<<(nW1 + 255) / 256, 256, 0, stream>>>(w3b, Aw3b, C, 27);

        const int NPOS = BSZ * HW;
        dim3 g3((HW + 63) / 64, C / 128, BSZ);
        dim3 g1((NPOS + 63) / 64);
        if (i == 0) {
            k_conv3x3_wmma<256, 80, 80><<<g3, 256, 0, stream>>>(in16, Aw2a, b2a, int2);
            k_conv3x3_wmma<256, 80, 80><<<g3, 256, 0, stream>>>(in16, Aw3a, b3a, int3);
            k_conv1x1_wmma<256><<<g1, 256, 0, stream>>>(int2, Aw2b, b2b, h2raw, HW, 24);
            k_conv1x1_wmma<256><<<g1, 256, 0, stream>>>(int3, Aw3b, b3b, h3raw, HW, 27);
        } else if (i == 1) {
            k_conv3x3_wmma<512, 40, 40><<<g3, 256, 0, stream>>>(in16, Aw2a, b2a, int2);
            k_conv3x3_wmma<512, 40, 40><<<g3, 256, 0, stream>>>(in16, Aw3a, b3a, int3);
            k_conv1x1_wmma<512><<<g1, 256, 0, stream>>>(int2, Aw2b, b2b, h2raw, HW, 24);
            k_conv1x1_wmma<512><<<g1, 256, 0, stream>>>(int3, Aw3b, b3b, h3raw, HW, 27);
        } else {
            k_conv3x3_wmma<1024, 20, 20><<<g3, 256, 0, stream>>>(in16, Aw2a, b2a, int2);
            k_conv3x3_wmma<1024, 20, 20><<<g3, 256, 0, stream>>>(in16, Aw3a, b3a, int3);
            k_conv1x1_wmma<1024><<<g1, 256, 0, stream>>>(int2, Aw2b, b2b, h2raw, HW, 24);
            k_conv1x1_wmma<1024><<<g1, 256, 0, stream>>>(int3, Aw3b, b3b, h3raw, HW, 27);
        }

        const int nR = BSZ * 3 * HW;
        k_decode<<<(nR + 255) / 256, 256, 0, stream>>>(
            h2raw, h3raw, (float*)d_out, HW, W, strides[i],
            anch[i][0], anch[i][1], anch[i][2], anch[i][3], anch[i][4], anch[i][5],
            scaleOff[i]);
    }
}
